// DifferentiableKMeans_17265768529988
// MI455X (gfx1250) — compile-verified, matrix operand order checked by reading, not run on hardware
//
#include <hip/hip_runtime.h>
#include <cstddef>

// Problem constants (match the reference).
constexpr int Nc = 32768;
constexpr int Dc = 768;
constexpr int Kc = 1024;

typedef __attribute__((ext_vector_type(16))) __bf16 v16bf;
typedef __attribute__((ext_vector_type(8)))  float  v8f;

// One WMMA fragment half-view: 16 bf16 = two 16-byte quads (b128 loads).
union Frag {
    v16bf v;
    uint4 q[2];
};

// ---------------------------------------------------------------------------
// Kernel 0: split fp32 -> (hi, lo) bf16 pair arrays (bf16x3 emulation data).
// Vectorized: each thread handles 4 consecutive floats.
// ---------------------------------------------------------------------------
__global__ void __launch_bounds__(256)
split_kernel(const float* __restrict__ src, __bf16* __restrict__ hi,
             __bf16* __restrict__ lo, int n4) {
    const int i = blockIdx.x * 256 + threadIdx.x;
    if (i >= n4) return;
    const float4 f = ((const float4*)src)[i];
    float v[4] = {f.x, f.y, f.z, f.w};
    union { __bf16 b[4]; uint2 u; } ph, pl;
#pragma unroll
    for (int j = 0; j < 4; ++j) {
        __bf16 h  = (__bf16)v[j];        // RTNE fp32 -> bf16
        float rsd = v[j] - (float)h;     // residual
        ph.b[j] = h;
        pl.b[j] = (__bf16)rsd;
    }
    ((uint2*)hi)[i] = ph.u;
    ((uint2*)lo)[i] = pl.u;
}

// ---------------------------------------------------------------------------
// Kernel 1: squared row norms for X (blocks [0,N)) and C (blocks [N,N+K)).
// ---------------------------------------------------------------------------
__global__ void __launch_bounds__(256)
sqnorm_kernel(const float* __restrict__ X, const float* __restrict__ Cm,
              float* __restrict__ x2, float* __restrict__ c2) {
    __shared__ float red[256];
    const int row = blockIdx.x;
    const float* p;
    float* out;
    int idx;
    if (row < Nc) { p = X  + (size_t)row * Dc;        out = x2; idx = row; }
    else          { p = Cm + (size_t)(row - Nc) * Dc; out = c2; idx = row - Nc; }

    float s = 0.f;
    for (int i = threadIdx.x; i < Dc; i += 256) {
        float v = p[i];
        s += v * v;
    }
    red[threadIdx.x] = s;
    __syncthreads();
#pragma unroll
    for (int off = 128; off > 0; off >>= 1) {
        if (threadIdx.x < off) red[threadIdx.x] += red[threadIdx.x + off];
        __syncthreads();
    }
    if (threadIdx.x == 0) out[idx] = red[0];
}

// ---------------------------------------------------------------------------
// Load one lane's half-fragment (16 bf16) from a pre-split row-major array.
// Lane (hlf, r) takes elements [d0, d0+8) and [d0+16, d0+24) of its row,
// matching the CDNA5 16-bit A-matrix layout (B assumed symmetric in N).
// ---------------------------------------------------------------------------
__device__ __forceinline__ v16bf load_half(const __bf16* __restrict__ row, int d0) {
    Frag f;
    f.q[0] = *(const uint4*)(row + d0);
    f.q[1] = *(const uint4*)(row + d0 + 16);
    return f.v;
}

// ---------------------------------------------------------------------------
// Kernel 2: WMMA GEMM  S = X * C^T  via bf16x3 (hi*hi + hi*lo + lo*hi, fp32
// accumulate), fused into d = sqrt(max(x2 + c2 - 2 s, 0)), stored as Dks[K][N].
// Wave tile = 32 points x 64 centroids (8 v8f accumulators).
// Block = 8 waves = 256 points x 64 centroids.
// ---------------------------------------------------------------------------
__global__ void __launch_bounds__(256)
gemm_dist_kernel(const __bf16* __restrict__ Xhi, const __bf16* __restrict__ Xlo,
                 const __bf16* __restrict__ Chi, const __bf16* __restrict__ Clo,
                 const float* __restrict__ x2, const float* __restrict__ c2,
                 float* __restrict__ Dks) {
    const int lane = threadIdx.x & 31;
    const int wave = threadIdx.x >> 5;
    const int hlf  = lane >> 4;       // which 16-lane half of the wave
    const int r    = lane & 15;

    const int m0 = blockIdx.y * 256 + wave * 32;  // point-tile base (2 sub-tiles)
    const int c0 = blockIdx.x * 64;               // centroid-tile base (4 sub-tiles)

    const __bf16* xh0 = Xhi + (size_t)(m0 + r) * Dc;
    const __bf16* xl0 = Xlo + (size_t)(m0 + r) * Dc;
    const __bf16* xh1 = xh0 + (size_t)16 * Dc;
    const __bf16* xl1 = xl0 + (size_t)16 * Dc;

    const __bf16* ch[4];
    const __bf16* cl[4];
#pragma unroll
    for (int j = 0; j < 4; ++j) {
        ch[j] = Chi + (size_t)(c0 + 16 * j + r) * Dc;
        cl[j] = Clo + (size_t)(c0 + 16 * j + r) * Dc;
    }

    v8f acc[8] = {};   // [j] = m-tile0, [4+j] = m-tile1

    for (int dbase = 0; dbase < Dc; dbase += 32) {
        const int d0 = dbase + hlf * 8;
        const v16bf a0h = load_half(xh0, d0);
        const v16bf a0l = load_half(xl0, d0);
        const v16bf a1h = load_half(xh1, d0);
        const v16bf a1l = load_half(xl1, d0);
#pragma unroll
        for (int j = 0; j < 4; ++j) {
            const v16bf bh = load_half(ch[j], d0);
            const v16bf bl = load_half(cl[j], d0);
            acc[j] = __builtin_amdgcn_wmma_f32_16x16x32_bf16(
                false, a0h, false, bh, (short)0, acc[j], false, false);
            acc[j] = __builtin_amdgcn_wmma_f32_16x16x32_bf16(
                false, a0h, false, bl, (short)0, acc[j], false, false);
            acc[j] = __builtin_amdgcn_wmma_f32_16x16x32_bf16(
                false, a0l, false, bh, (short)0, acc[j], false, false);
            acc[4 + j] = __builtin_amdgcn_wmma_f32_16x16x32_bf16(
                false, a1h, false, bh, (short)0, acc[4 + j], false, false);
            acc[4 + j] = __builtin_amdgcn_wmma_f32_16x16x32_bf16(
                false, a1h, false, bl, (short)0, acc[4 + j], false, false);
            acc[4 + j] = __builtin_amdgcn_wmma_f32_16x16x32_bf16(
                false, a1l, false, bh, (short)0, acc[4 + j], false, false);
        }
    }

    // Epilogue. C/D layout: lane holds column n = r (centroid offset),
    // VGPR rr holds row m = rr + 8*hlf (point offset within 16-row tile).
#pragma unroll
    for (int mt = 0; mt < 2; ++mt) {
        const int mb = m0 + mt * 16 + hlf * 8;
#pragma unroll
        for (int j = 0; j < 4; ++j) {
            const int c  = c0 + 16 * j + r;
            const float cc = c2[c];
            float tmp[8];
#pragma unroll
            for (int rr = 0; rr < 8; ++rr) {
                float s  = acc[4 * mt + j][rr];
                float d2 = x2[mb + rr] + cc - 2.0f * s;
                tmp[rr]  = sqrtf(fmaxf(d2, 0.0f));
            }
            float* drow = Dks + (size_t)c * Nc + mb;
            *(float4*)(drow)     = make_float4(tmp[0], tmp[1], tmp[2], tmp[3]);
            *(float4*)(drow + 4) = make_float4(tmp[4], tmp[5], tmp[6], tmp[7]);
        }
    }
}

// ---------------------------------------------------------------------------
// Kernel 3: per-point argmin over K (coalesced reads of Dks[K][N]).
// Also zeroes the loss accumulator d_out[0] (next kernel atomically adds).
// ---------------------------------------------------------------------------
__global__ void __launch_bounds__(256)
argmin_kernel(const float* __restrict__ Dks, float* __restrict__ out) {
    const int n = blockIdx.x * 256 + threadIdx.x;
    float best = 3.402823466e+38f;
    int   bi   = 0;
    for (int k = 0; k < Kc; ++k) {
        float d = Dks[(size_t)k * Nc + n];
        if (d < best) { best = d; bi = k; }
    }
    out[1 + n] = (float)bi;
    if (blockIdx.x == 0 && threadIdx.x == 0) out[0] = 0.0f;
}

// ---------------------------------------------------------------------------
// Kernel 4: per-centroid stable softmin over the N axis; block per centroid.
// loss_k = sum_n d*exp(min-d) / sum_n exp(min-d);  out[0] += loss_k / K.
// ---------------------------------------------------------------------------
__global__ void __launch_bounds__(256)
softmin_kernel(const float* __restrict__ Dks, float* __restrict__ out) {
    __shared__ float redA[256];
    __shared__ float redB[256];
    const int k = blockIdx.x;
    const int t = threadIdx.x;
    const float* row = Dks + (size_t)k * Nc;

    // Pass 1: min over the row.
    float m = 3.402823466e+38f;
    for (int i = t; i < Nc; i += 256) m = fminf(m, row[i]);
    redA[t] = m;
    __syncthreads();
#pragma unroll
    for (int off = 128; off > 0; off >>= 1) {
        if (t < off) redA[t] = fminf(redA[t], redA[t + off]);
        __syncthreads();
    }
    const float mn = redA[0];
    __syncthreads();

    // Pass 2: sums of exp(mn - d) and d*exp(mn - d)   (ALPHA = 1).
    float s = 0.f, ws = 0.f;
    for (int i = t; i < Nc; i += 256) {
        float d = row[i];
        float e = __expf(mn - d);
        s  += e;
        ws += d * e;
    }
    redA[t] = s;
    redB[t] = ws;
    __syncthreads();
#pragma unroll
    for (int off = 128; off > 0; off >>= 1) {
        if (t < off) {
            redA[t] += redA[t + off];
            redB[t] += redB[t + off];
        }
        __syncthreads();
    }
    if (t == 0) atomicAdd(out, (redB[0] / redA[0]) * (1.0f / (float)Kc));
}

// ---------------------------------------------------------------------------
// Host launcher.
// ---------------------------------------------------------------------------
extern "C" void kernel_launch(void* const* d_in, const int* in_sizes, int n_in,
                              void* d_out, int out_size, void* d_ws, size_t ws_size,
                              hipStream_t stream) {
    (void)in_sizes; (void)n_in; (void)out_size; (void)ws_size;

    const float* X  = (const float*)d_in[0];   // [N, D] fp32
    const float* Cm = (const float*)d_in[1];   // [K, D] fp32
    float* out = (float*)d_out;                // [0]=loss, [1..N]=argmin (as float)

    // Workspace layout (floats unless noted):
    //   Dks[K*N] | x2[N] | c2[K] | Xhi,Xlo (N*D bf16) | Chi,Clo (K*D bf16)
    float* Dks = (float*)d_ws;
    float* x2  = Dks + (size_t)Kc * Nc;
    float* c2  = x2 + Nc;
    __bf16* Xhi = (__bf16*)(c2 + Kc);
    __bf16* Xlo = Xhi + (size_t)Nc * Dc;
    __bf16* Chi = Xlo + (size_t)Nc * Dc;
    __bf16* Clo = Chi + (size_t)Kc * Dc;

    const int xn4 = Nc * Dc / 4;
    const int cn4 = Kc * Dc / 4;

    split_kernel<<<(xn4 + 255) / 256, 256, 0, stream>>>(X, Xhi, Xlo, xn4);
    split_kernel<<<(cn4 + 255) / 256, 256, 0, stream>>>(Cm, Chi, Clo, cn4);
    sqnorm_kernel<<<Nc + Kc, 256, 0, stream>>>(X, Cm, x2, c2);
    gemm_dist_kernel<<<dim3(Kc / 64, Nc / 256), 256, 0, stream>>>(
        Xhi, Xlo, Chi, Clo, x2, c2, Dks);
    argmin_kernel<<<Nc / 256, 256, 0, stream>>>(Dks, out);
    softmin_kernel<<<Kc, 256, 0, stream>>>(Dks, out);
}
